// VisionTransformer_3934190044250
// MI455X (gfx1250) — matrix operand order repaired and checked by reading, not compiled, not run on hardware
//
#include <hip/hip_runtime.h>
#include <math.h>

// ---------------------------------------------------------------------------
// ViT-B/16 + adapters forward for MI455X (gfx1250, wave32, WMMA).
// All GEMMs go through v_wmma_f32_16x16x32_f16 (fp32 staged->f16 in LDS,
// f32 accumulate). Sequence length padded 197 -> 224 (multiple of 32).
// Staging: per-thread contiguous half-row -> 4x global_load_b128,
// 8x v_cvt_pk_f16_f32, 2x ds_store_b128 (latency batched, no exec-mask loops).
// ---------------------------------------------------------------------------

typedef _Float16     half16  __attribute__((ext_vector_type(16)));
typedef float        float8  __attribute__((ext_vector_type(8)));
typedef float        float4v __attribute__((ext_vector_type(4)));
typedef unsigned int uint4v  __attribute__((ext_vector_type(4)));

union FragH { half16 h; unsigned int u[8]; };

#define LP   224          // padded sequence length (multiple of 32 and 16)
#define LR   197          // real sequence length
#define DM   768
#define NH   12
#define DH   64
#define BB   8

static __device__ __forceinline__ unsigned int pk2(float x, float y)
{
    union { _Float16 h[2]; unsigned int u; } w;
    w.h[0] = (_Float16)x; w.h[1] = (_Float16)y;   // -> v_cvt_pk_f16_f32
    return w.u;
}

static __device__ __forceinline__ void cvt_store8(const float4v f[4], unsigned int* dst)
{
    uint4v u0, u1;
    u0[0] = pk2(f[0][0], f[0][1]); u0[1] = pk2(f[0][2], f[0][3]);
    u0[2] = pk2(f[1][0], f[1][1]); u0[3] = pk2(f[1][2], f[1][3]);
    u1[0] = pk2(f[2][0], f[2][1]); u1[1] = pk2(f[2][2], f[2][3]);
    u1[2] = pk2(f[3][0], f[3][1]); u1[3] = pk2(f[3][2], f[3][3]);
    *(uint4v*)dst       = u0;                      // ds_store_b128
    *(uint4v*)(dst + 4) = u1;                      // ds_store_b128
}

// ---------------------------------------------------------------------------
// Generic tiled WMMA GEMM:  C[m,n] = sum_k A[m,k] * B'[k,n]  (+ epilogue)
//   BNT=true : B given as N x K row-major (weights, x @ W^T)
//   BNT=false: B given as K x N row-major (att @ V, y @ proj)
// EPI: 0 none | 1 +bias | 2 +bias,quickgelu | 3 +bias,gelu | 4 +bias,+R
// grid.z = B1*Hn with per-level batch strides (attention: per-(b,h) slices).
// Block = 128 threads = 4 wave32s; each wave owns a 32x32 C sub-tile
// (2x2 of 16x16x32 WMMA); block tile = 64x64, K-slab = 32.
// ---------------------------------------------------------------------------
template<bool BNT, int EPI>
__global__ void __launch_bounds__(128) gemm_wmma(
    const float* __restrict__ A, const float* __restrict__ Bm,
    const float* __restrict__ bias, const float* __restrict__ R,
    float* __restrict__ C,
    int M, int N, int K, int lda, int ldb, int ldc,
    long long aB, long long aH, long long bB, long long bH,
    long long cB, long long cH, int Hn)
{
    __shared__ unsigned int sA[64 * 16];   // 64 rows x 16 f16-pairs (K=32)
    __shared__ unsigned int sB[64 * 16];   // 64 cols x 16 f16-pairs, [n][k]

    const int z  = blockIdx.z;
    const int b1 = z / Hn;
    const int hh = z - b1 * Hn;
    A  += (long long)b1 * aB + (long long)hh * aH;
    Bm += (long long)b1 * bB + (long long)hh * bH;
    C  += (long long)b1 * cB + (long long)hh * cH;
    const float* Rm = R;
    if (EPI == 4) Rm = R + (long long)b1 * cB + (long long)hh * cH;

    const int tid  = threadIdx.x;
    const int lane = tid & 31;
    const int wave = tid >> 5;
    const int mTile = blockIdx.y * 64;
    const int nTile = blockIdx.x * 64;
    const int wm = (wave >> 1) * 32;       // wave's 32x32 sub-tile
    const int wn = (wave & 1) * 32;

    // staging coordinates: thread owns row/col sr, K-half sh (16 floats)
    const int sr  = tid >> 1;              // 0..63
    const int sh  = tid & 1;               // 0 -> k 0..15, 1 -> k 16..31
    const int gmS = mTile + sr;
    const int gnS = nTile + sr;
    unsigned int* dstA = &sA[(sr << 4) + sh * 8];
    unsigned int* dstB = &sB[(sr << 4) + sh * 8];

    float8 acc[2][2];
    #pragma unroll
    for (int i = 0; i < 2; ++i)
        #pragma unroll
        for (int j = 0; j < 2; ++j)
            acc[i][j] = (float8)0.0f;

    for (int kb = 0; kb < K; kb += 32) {
        // ---- stage A half-row: 4x b128 load, 8x cvt_pk, 2x ds b128 ----
        {
            float4v f[4] = {(float4v)0.f, (float4v)0.f, (float4v)0.f, (float4v)0.f};
            if (gmS < M) {
                const float* ap = A + (long long)gmS * lda + kb + sh * 16;
                #pragma unroll
                for (int q = 0; q < 4; ++q) f[q] = *(const float4v*)(ap + 4 * q);
            }
            cvt_store8(f, dstA);
        }
        // ---- stage B half-row / half-column ----
        {
            float4v f[4] = {(float4v)0.f, (float4v)0.f, (float4v)0.f, (float4v)0.f};
            if (gnS < N) {
                if (BNT) {
                    const float* bp = Bm + (long long)gnS * ldb + kb + sh * 16;
                    #pragma unroll
                    for (int q = 0; q < 4; ++q) f[q] = *(const float4v*)(bp + 4 * q);
                } else {
                    const float* bp = Bm + (long long)(kb + sh * 16) * ldb + gnS;
                    #pragma unroll
                    for (int q = 0; q < 16; ++q)
                        f[q >> 2][q & 3] = bp[(long long)q * ldb];
                }
            }
            cvt_store8(f, dstB);
        }
        __syncthreads();

        // prefetch next A K-slab into cache (global_prefetch_b8)
        if (kb + 32 < K && sh == 0 && gmS < M)
            __builtin_prefetch(A + (long long)gmS * lda + kb + 32, 0, 3);

        // ---- build fragments, contiguous k = 16*half + s for A and B alike (wave32) ----
        const int half = lane >> 4;
        const int l16  = lane & 15;
        FragH fa[2], fb[2];
        #pragma unroll
        for (int i = 0; i < 2; ++i) {
            int row = wm + i * 16 + l16;
            #pragma unroll
            for (int v = 0; v < 8; ++v) {
                // A 16x32 f16: half0 -> K pairs 0..7, half1 -> 8..15
                int p2 = v + half * 8;
                fa[i].u[v] = sA[(row << 4) + p2];
            }
        }
        #pragma unroll
        for (int j = 0; j < 2; ++j) {
            int col = wn + j * 16 + l16;
            #pragma unroll
            for (int v = 0; v < 8; ++v) {
                // B 32x16 f16: half0 -> K pairs 0..7, half1 -> 8..15
                fb[j].u[v] = sB[(col << 4) + v + half * 8];
            }
        }
        #pragma unroll
        for (int i = 0; i < 2; ++i)
            #pragma unroll
            for (int j = 0; j < 2; ++j)
                acc[i][j] = __builtin_amdgcn_wmma_f32_16x16x32_f16(
                    false, fa[i].h, false, fb[j].h,
                    (short)0, acc[i][j], false, false);
        __syncthreads();
    }

    // ---- epilogue + store (C/D layout: lane half selects M+8) ----
    const int half = lane >> 4;
    const int l16  = lane & 15;
    #pragma unroll
    for (int i = 0; i < 2; ++i) {
        #pragma unroll
        for (int j = 0; j < 2; ++j) {
            #pragma unroll
            for (int v = 0; v < 8; ++v) {
                int row = mTile + wm + i * 16 + v + half * 8;
                int col = nTile + wn + j * 16 + l16;
                if (row < M && col < N) {
                    float val = acc[i][j][v];
                    if (EPI >= 1) val += bias[col];
                    if (EPI == 2) val = val * (1.f / (1.f + __expf(-1.702f * val)));
                    if (EPI == 3) val = 0.5f * val * (1.f + erff(val * 0.70710678f));
                    if (EPI == 4) val += Rm[(long long)row * ldc + col];
                    C[(long long)row * ldc + col] = val;
                }
            }
        }
    }
}

// ---------------------------------------------------------------------------
// Patchify: x (B,3,224,224) -> patches (B, 196(->pad 224), 768) in (c,py,px)
// order, matching conv_w.reshape(768, 768) rows.
// ---------------------------------------------------------------------------
__global__ void __launch_bounds__(256) patchify_kernel(
    const float* __restrict__ x, float* __restrict__ patches)
{
    int idx = blockIdx.x * 256 + threadIdx.x;
    const int total = BB * 196 * DM;
    if (idx >= total) return;
    int b   = idx / (196 * DM);
    int rem = idx - b * 196 * DM;
    int g   = rem / DM;
    int ci  = rem - g * DM;
    int c   = ci >> 8;               // /256
    int pr  = (ci & 255) >> 4;
    int pc  = ci & 15;
    int gy  = g / 14, gx = g - gy * 14;
    float v = x[(((long long)b * 3 + c) * 224 + gy * 16 + pr) * 224 + gx * 16 + pc];
    patches[((long long)b * LP + g) * DM + ci] = v;
}

// cls token + positional embedding (rows 1..196 already hold patch embeds)
__global__ void __launch_bounds__(256) addcls_kernel(
    float* __restrict__ xv, const float* __restrict__ cls,
    const float* __restrict__ pos)
{
    int idx = blockIdx.x * 256 + threadIdx.x;
    const int total = BB * LR * DM;
    if (idx >= total) return;
    int b   = idx / (LR * DM);
    int rem = idx - b * LR * DM;
    int r   = rem / DM;
    int d   = rem - r * DM;
    float* p = xv + ((long long)b * LP + r) * DM + d;
    if (r == 0) *p = cls[d] + pos[d];
    else        *p = *p + pos[(long long)r * DM + d];
}

// row LayerNorm over D=768 (256 threads x 3 elements)
__global__ void __launch_bounds__(256) ln_kernel(
    const float* __restrict__ in, float* __restrict__ out,
    const float* __restrict__ gg, const float* __restrict__ bb,
    int rowsTotal, int rpb, long long bsIn, long long bsOut)
{
    __shared__ float red[256];
    __shared__ float s_mu, s_rstd;
    int r = blockIdx.x;
    if (r >= rowsTotal) return;
    int b = r / rpb, i = r - b * rpb;
    const float* p = in  + (long long)b * bsIn  + (long long)i * DM;
    float*       q = out + (long long)b * bsOut + (long long)i * DM;
    int t = threadIdx.x;
    float v0 = p[t], v1 = p[t + 256], v2 = p[t + 512];
    red[t] = v0 + v1 + v2;
    __syncthreads();
    for (int s = 128; s > 0; s >>= 1) { if (t < s) red[t] += red[t + s]; __syncthreads(); }
    if (t == 0) s_mu = red[0] * (1.f / 768.f);
    __syncthreads();
    float mu = s_mu;
    float d0 = v0 - mu, d1 = v1 - mu, d2 = v2 - mu;
    red[t] = d0 * d0 + d1 * d1 + d2 * d2;
    __syncthreads();
    for (int s = 128; s > 0; s >>= 1) { if (t < s) red[t] += red[t + s]; __syncthreads(); }
    if (t == 0) s_rstd = rsqrtf(red[0] * (1.f / 768.f) + 1e-5f);
    __syncthreads();
    float rs = s_rstd;
    q[t]       = d0 * rs * gg[t]       + bb[t];
    q[t + 256] = d1 * rs * gg[t + 256] + bb[t + 256];
    q[t + 512] = d2 * rs * gg[t + 512] + bb[t + 512];
}

// masked softmax over padded attention rows (length 224, valid 197), with QK scale
__global__ void __launch_bounds__(256) softmax_kernel(
    float* __restrict__ att, int rowsTotal, float scale)
{
    __shared__ float red[256];
    __shared__ float s_val;
    int r = blockIdx.x;
    if (r >= rowsTotal) return;
    int bh = r / LR, i = r - bh * LR;
    float* p = att + (long long)bh * LP * LP + (long long)i * LP;
    int t = threadIdx.x;
    float v = (t < LR) ? p[t] * scale : -1e30f;
    red[t] = v;
    __syncthreads();
    for (int s = 128; s > 0; s >>= 1) { if (t < s) red[t] = fmaxf(red[t], red[t + s]); __syncthreads(); }
    if (t == 0) s_val = red[0];
    __syncthreads();
    float mx = s_val;
    float e = (t < LR) ? __expf(v - mx) : 0.f;
    red[t] = e;
    __syncthreads();
    for (int s = 128; s > 0; s >>= 1) { if (t < s) red[t] += red[t + s]; __syncthreads(); }
    if (t == 0) s_val = red[0];
    __syncthreads();
    float inv = 1.f / s_val;
    if (t < LP) p[t] = (t < LR) ? e * inv : 0.f;
}

// ---------------------------------------------------------------------------
// Host-side dispatch
// ---------------------------------------------------------------------------
static void run_gemm(hipStream_t s, bool bnt, int epi,
                     const float* A, const float* Bm, const float* bias,
                     const float* R, float* C,
                     int M, int N, int K, int lda, int ldb, int ldc,
                     int B1, int Hn,
                     long long aB, long long aH, long long bB, long long bH,
                     long long cB, long long cH)
{
    dim3 g((N + 63) / 64, (M + 63) / 64, B1 * Hn), b(128);
#define GARGS A, Bm, bias, R, C, M, N, K, lda, ldb, ldc, aB, aH, bB, bH, cB, cH, Hn
    if (bnt) {
        switch (epi) {
        case 0: gemm_wmma<true, 0><<<g, b, 0, s>>>(GARGS); break;
        case 1: gemm_wmma<true, 1><<<g, b, 0, s>>>(GARGS); break;
        case 2: gemm_wmma<true, 2><<<g, b, 0, s>>>(GARGS); break;
        case 3: gemm_wmma<true, 3><<<g, b, 0, s>>>(GARGS); break;
        default: gemm_wmma<true, 4><<<g, b, 0, s>>>(GARGS); break;
        }
    } else {
        gemm_wmma<false, 0><<<g, b, 0, s>>>(GARGS);
    }
#undef GARGS
}

extern "C" void kernel_launch(void* const* d_in, const int* in_sizes, int n_in,
                              void* d_out, int out_size, void* d_ws, size_t ws_size,
                              hipStream_t stream)
{
    (void)in_sizes; (void)n_in; (void)out_size;
    const float* x_in    = (const float*)d_in[0];
    const float* conv_w  = (const float*)d_in[1];
    const float* cls_emb = (const float*)d_in[2];
    const float* pos_emb = (const float*)d_in[3];
    const float* lnpre_g = (const float*)d_in[4];
    const float* lnpre_b = (const float*)d_in[5];
    const float* in_w    = (const float*)d_in[6];
    const float* in_b    = (const float*)d_in[7];
    const float* out_w   = (const float*)d_in[8];
    const float* out_b   = (const float*)d_in[9];
    const float* ln1_g   = (const float*)d_in[10];
    const float* ln1_b   = (const float*)d_in[11];
    const float* fc_w    = (const float*)d_in[12];
    const float* fc_b    = (const float*)d_in[13];
    const float* pj_w    = (const float*)d_in[14];
    const float* pj_b    = (const float*)d_in[15];
    const float* ln2_g   = (const float*)d_in[16];
    const float* ln2_b   = (const float*)d_in[17];
    const float* ad_lng  = (const float*)d_in[18];
    const float* ad_lnb  = (const float*)d_in[19];
    const float* ad_dw   = (const float*)d_in[20];
    const float* ad_db   = (const float*)d_in[21];
    const float* ad_uw   = (const float*)d_in[22];
    const float* ad_ub   = (const float*)d_in[23];
    const float* lnpost_g = (const float*)d_in[24];
    const float* lnpost_b = (const float*)d_in[25];
    const float* proj     = (const float*)d_in[26];
    float* out = (float*)d_out;

    float* ws = (float*)d_ws;
    // workspace layout (floats)
    size_t off = 0;
    float* patches = ws + off; off += (size_t)BB * LP * DM;        // 8x224x768
    float* xbuf    = ws + off; off += (size_t)BB * LP * DM;
    float* ybuf    = ws + off; off += (size_t)BB * LP * DM;
    float* qkv     = ws + off; off += (size_t)BB * LP * 3 * DM;    // 8x224x2304
    float* att     = ws + off; off += (size_t)BB * NH * LP * LP;   // 96x224x224
    float* obuf    = ws + off; off += (size_t)BB * LP * DM;
    float* hbuf    = ws + off; off += (size_t)BB * LP * 4 * DM;    // 8x224x3072
    float* abuf    = ws + off; off += (size_t)BB * LP * 192;
    float* clsbuf  = ws + off; off += (size_t)BB * DM;
    size_t need = off * sizeof(float);
    hipMemsetAsync(d_ws, 0, need <= ws_size ? need : ws_size, stream);

    const long long XB = (long long)LP * DM;        // x/y/o batch stride
    const long long QB = (long long)LP * 3 * DM;    // qkv batch stride
    const long long HB = (long long)LP * 4 * DM;    // mlp hidden batch stride
    const long long AB = (long long)NH * LP * LP;   // att per-image stride

    // --- patch embed ---
    {
        int total = BB * 196 * DM;
        patchify_kernel<<<(total + 255) / 256, 256, 0, stream>>>(x_in, patches);
        // xv rows 1..196 = patches @ conv_w^T
        run_gemm(stream, true, 0, patches, conv_w, nullptr, nullptr, xbuf + DM,
                 196, DM, DM, DM, DM, DM, BB, 1, XB, 0, 0, 0, XB, 0);
        int tot2 = BB * LR * DM;
        addcls_kernel<<<(tot2 + 255) / 256, 256, 0, stream>>>(xbuf, cls_emb, pos_emb);
        ln_kernel<<<BB * LR, 256, 0, stream>>>(xbuf, xbuf, lnpre_g, lnpre_b,
                                               BB * LR, LR, XB, XB);
    }

    const float scale = 0.125f;  // 64^-0.5

    for (int j = 0; j < 11; ++j) {
        // ---- attention ----
        ln_kernel<<<BB * LR, 256, 0, stream>>>(xbuf, ybuf, ln1_g + j * DM, ln1_b + j * DM,
                                               BB * LR, LR, XB, XB);
        // qkv = y @ in_w^T + in_b
        run_gemm(stream, true, 1, ybuf, in_w + (size_t)j * 3 * DM * DM,
                 in_b + (size_t)j * 3 * DM, nullptr, qkv,
                 LR, 3 * DM, DM, DM, DM, 3 * DM, BB, 1, XB, 0, 0, 0, QB, 0);
        // att = Q @ K^T  (per (b,h))
        run_gemm(stream, true, 0, qkv, qkv + DM, nullptr, nullptr, att,
                 LR, LR, DH, 3 * DM, 3 * DM, LP, BB, NH,
                 QB, DH, QB, DH, AB, (long long)LP * LP);
        softmax_kernel<<<BB * NH * LR, 256, 0, stream>>>(att, BB * NH * LR, scale);
        // o = att @ V  (per (b,h)), K padded to 224 (extra cols are zero)
        run_gemm(stream, false, 0, att, qkv + 2 * DM, nullptr, nullptr, obuf,
                 LR, DH, LP, LP, 3 * DM, DM, BB, NH,
                 AB, (long long)LP * LP, QB, DH, XB, DH);
        // x = x + o @ out_w^T + out_b
        run_gemm(stream, true, 4, obuf, out_w + (size_t)j * DM * DM,
                 out_b + (size_t)j * DM, xbuf, xbuf,
                 LR, DM, DM, DM, DM, DM, BB, 1, XB, 0, 0, 0, XB, 0);
        // ---- MLP ----
        ln_kernel<<<BB * LR, 256, 0, stream>>>(xbuf, ybuf, ln2_g + j * DM, ln2_b + j * DM,
                                               BB * LR, LR, XB, XB);
        run_gemm(stream, true, 2, ybuf, fc_w + (size_t)j * 4 * DM * DM,
                 fc_b + (size_t)j * 4 * DM, nullptr, hbuf,
                 LR, 4 * DM, DM, DM, DM, 4 * DM, BB, 1, XB, 0, 0, 0, HB, 0);
        run_gemm(stream, true, 4, hbuf, pj_w + (size_t)j * DM * 4 * DM,
                 pj_b + (size_t)j * DM, xbuf, xbuf,
                 LR, DM, 4 * DM, 4 * DM, 4 * DM, DM, BB, 1, HB, 0, 0, 0, XB, 0);

        // ---- adapter after last block of each group (j = 1,4,7,10) ----
        int g = -1;
        if (j == 1) g = 0; else if (j == 4) g = 1; else if (j == 7) g = 2; else if (j == 10) g = 3;
        if (g >= 0) {
            ln_kernel<<<BB * LR, 256, 0, stream>>>(xbuf, ybuf, ad_lng + g * DM, ad_lnb + g * DM,
                                                   BB * LR, LR, XB, XB);
            run_gemm(stream, true, 3, ybuf, ad_dw + (size_t)g * 192 * DM,
                     ad_db + (size_t)g * 192, nullptr, abuf,
                     LR, 192, DM, DM, DM, 192, BB, 1, XB, 0, 0, 0, (long long)LP * 192, 0);
            run_gemm(stream, true, 4, abuf, ad_uw + (size_t)g * DM * 192,
                     ad_ub + (size_t)g * DM, xbuf, xbuf,
                     LR, DM, 192, 192, 192, DM, BB, 1, (long long)LP * 192, 0, 0, 0, XB, 0);
        }
    }

    // ---- head: ln_post on CLS rows, then @ proj (768x512, K x N) ----
    ln_kernel<<<BB, 256, 0, stream>>>(xbuf, clsbuf, lnpost_g, lnpost_b,
                                      BB, 1, XB, (long long)DM);
    run_gemm(stream, false, 0, clsbuf, proj, nullptr, nullptr, out,
             BB, 512, DM, DM, 512, 512, 1, 1, 0, 0, 0, 0, 0, 0);
}
